// MoEFFN_10986526343382
// MI455X (gfx1250) — compile-verified, matrix operand order checked by reading, not run on hardware
//
#include <hip/hip_runtime.h>

// ---------------- problem constants ----------------
#define T_TOK 4096          // B*S tokens
#define DIM   768           // hidden dim D
#define FF    1536          // expert ffn dim F
#define NEXP  8             // routed experts
#define DK    (DIM/32)      // 24 k-tiles along D
#define FK    (FF/32)       // 48 k-tiles along F
#define NGRP  3             // expert groups (3+3+3 incl. shared)
#define HSTR  136           // Hs row stride in u16 (128 + 8 pad, bank-rotating)

typedef __bf16 v16bf __attribute__((ext_vector_type(16)));
typedef float  v8f   __attribute__((ext_vector_type(8)));
typedef unsigned short u16;
typedef unsigned int   u32;

static __device__ __forceinline__ u16 f2bf(float f) {
  u32 x = __float_as_uint(f);
  x += 0x7FFFu + ((x >> 16) & 1u);          // round-to-nearest-even
  return (u16)(x >> 16);
}

// =====================================================================
// Pack a row-major (Nrows x Kcols) fp32 matrix into bf16 WMMA A/B
// fragment tiles. Tile (nt,kt) -> 512 bf16, lane L holds 16 values:
//   n = nt*16 + (L&15)
//   k = kt*32 + { L<16 ? {0..7,16..23} : {8..15,24..31} }
// tileIdx = nt*(Kcols/32) + kt ; lane data at tileIdx*512 + L*16.
// =====================================================================
__global__ void moe_pack_frag(const float* __restrict__ W,
                              u16* __restrict__ out,
                              int Nrows, int Kcols) {
  int tile = blockIdx.x * 8 + (threadIdx.x >> 5);
  int lane = threadIdx.x & 31;
  int ktiles = Kcols >> 5;
  int nTiles = (Nrows >> 4) * ktiles;
  if (tile >= nTiles) return;
  int nt = tile / ktiles;
  int kt = tile - nt * ktiles;
  int n = nt * 16 + (lane & 15);
  int kbase = kt * 32 + ((lane >> 4) << 3);   // +8 for lanes 16..31
  const float* src = W + (size_t)n * Kcols + kbase;
  union { u16 u[16]; uint4 q[2]; } tmp;
#pragma unroll
  for (int s = 0; s < 8; s++) tmp.u[s]     = f2bf(src[s]);
#pragma unroll
  for (int s = 0; s < 8; s++) tmp.u[8 + s] = f2bf(src[16 + s]);
  uint4* dst = (uint4*)(out + (size_t)tile * 512 + lane * 16);
  dst[0] = tmp.q[0];
  dst[1] = tmp.q[1];
}

// =====================================================================
// Router: one wave per token. logits = x . w_router^T (fp32), softmax,
// biased top-2 selection, normalized weights -> comb[t][0..7],
// comb[t][8] = 1.0 (shared expert).
// =====================================================================
__global__ void moe_router(const float* __restrict__ x,
                           const float* __restrict__ wr,
                           const float* __restrict__ bias,
                           float* __restrict__ comb) {
  int wv = (blockIdx.x * blockDim.x + threadIdx.x) >> 5;   // token id
  if (wv >= T_TOK) return;
  int lane = threadIdx.x & 31;

  float acc[NEXP];
#pragma unroll
  for (int e = 0; e < NEXP; e++) acc[e] = 0.0f;

  for (int d = lane; d < DIM; d += 32) {
    float xv = x[(size_t)wv * DIM + d];
#pragma unroll
    for (int e = 0; e < NEXP; e++) acc[e] += xv * wr[(size_t)e * DIM + d];
  }
#pragma unroll
  for (int e = 0; e < NEXP; e++) {
#pragma unroll
    for (int off = 16; off >= 1; off >>= 1)
      acc[e] += __shfl_xor(acc[e], off, 32);
  }
  if (lane == 0) {
    float m = acc[0];
#pragma unroll
    for (int e = 1; e < NEXP; e++) m = fmaxf(m, acc[e]);
    float p[NEXP], s = 0.0f;
#pragma unroll
    for (int e = 0; e < NEXP; e++) { p[e] = __expf(acc[e] - m); s += p[e]; }
    float inv = 1.0f / s;
#pragma unroll
    for (int e = 0; e < NEXP; e++) p[e] *= inv;
    // biased top-2 (bias affects selection only)
    int i1 = 0; float b1 = p[0] + bias[0];
#pragma unroll
    for (int e = 1; e < NEXP; e++) {
      float v = p[e] + bias[e];
      if (v > b1) { b1 = v; i1 = e; }
    }
    int i2 = (i1 == 0) ? 1 : 0; float b2 = p[i2] + bias[i2];
#pragma unroll
    for (int e = 0; e < NEXP; e++) {
      if (e == i1) continue;
      float v = p[e] + bias[e];
      if (v > b2) { b2 = v; i2 = e; }
    }
    float s1 = p[i1], s2 = p[i2];
    float nrm = 1.0f / (s1 + s2 + 1e-9f);
    float* c = comb + (size_t)wv * 9;
#pragma unroll
    for (int e = 0; e < NEXP; e++) c[e] = 0.0f;
    c[i1] = s1 * nrm;
    c[i2] = s2 * nrm;
    c[8]  = 1.0f;   // shared expert always on
  }
}

// =====================================================================
// Fused MoE FFN: 16 waves / block (512 threads), 32 tokens / block.
// grid.y = expert group (3 experts each; group 2 includes the shared
// expert). The x tile (48 KB) is staged once in LDS: A-fragments are
// re-read from LDS each K-step, which (a) removes the 8x duplicate
// global loads across waves and (b) keeps the fragments out of long
// live ranges so the allocator cannot spill them. comb weight is folded
// into H before the down projection, so one fp32 accumulator set
// (6 tiles x 8 = 48 VGPRs) carries the whole group.
// =====================================================================
__global__ __launch_bounds__(512, 1)
void moe_main(const u16* __restrict__ xb,
              const u16* __restrict__ wgb, const u16* __restrict__ wub,
              const u16* __restrict__ wdb,
              const u16* __restrict__ sgb, const u16* __restrict__ sub,
              const u16* __restrict__ sdb,
              const float* __restrict__ comb,
              float* __restrict__ part) {
  __shared__ u16 Xs[2 * DK * 512];     // x fragments: 2 m-tiles x 24 k-tiles, 48 KB
  __shared__ u16 Hs[32 * HSTR];        // H chunk: 32 tokens x 128 cols (+pad), 8.5 KB

  const int wave  = threadIdx.x >> 5;
  const int lane  = threadIdx.x & 31;
  const int mhalf = wave & 1;          // which 16-token half
  const int wq    = wave >> 1;         // 0..7 eighth
  const int mtile = blockIdx.x * 2 + mhalf;
  const int tokBase = mtile * 16;
  const int rowSel = lane >> 4;        // 0/1 lane half
  const int nIdx   = lane & 15;
  const int grp    = blockIdx.y;       // expert group 0..2

  // ---- stage x tile into LDS (once per block) ----
  {
    const uint4* src = (const uint4*)(xb + (size_t)blockIdx.x * 2 * DK * 512);
    uint4* dst = (uint4*)Xs;
#pragma unroll
    for (int it = 0; it < 6; it++)
      dst[threadIdx.x + it * 512] = src[threadIdx.x + it * 512];
  }
  __syncthreads();

  v8f yacc[6];
#pragma unroll
  for (int j = 0; j < 6; j++)
#pragma unroll
    for (int i = 0; i < 8; i++) yacc[j][i] = 0.0f;

  const u16* xlds = Xs + (size_t)mhalf * DK * 512 + lane * 16;

  for (int ee = 0; ee < 3; ee++) {
    const int e = grp * 3 + ee;        // 0..8 ; 8 == shared expert
    const u16* wg = (e < NEXP) ? wgb + (size_t)e * FF * DIM : sgb;
    const u16* wu = (e < NEXP) ? wub + (size_t)e * FF * DIM : sub;
    const u16* wd = (e < NEXP) ? wdb + (size_t)e * DIM * FF : sdb;

    float cw[8];
#pragma unroll
    for (int i = 0; i < 8; i++)
      cw[i] = comb[(size_t)(tokBase + i + 8 * rowSel) * 9 + e];

    for (int ch = 0; ch < FF / 128; ch++) {    // 12 chunks of 128 f-cols
      // ---- stage 1: gate & up tiles (16 tokens x 16 f-cols each) ----
      const int ftile = ch * 8 + wq;           // 0..95 within expert
      v8f g, u;
#pragma unroll
      for (int i = 0; i < 8; i++) { g[i] = 0.0f; u[i] = 0.0f; }
      const u16* wgp = wg + (size_t)ftile * DK * 512 + lane * 16;
      const u16* wup = wu + (size_t)ftile * DK * 512 + lane * 16;
      const u16* xp  = xlds;
      for (int kt = 0; kt < DK; kt++) {
        union { uint4 q[2]; v16bf v; } ax;
        ax.q[0] = *(const uint4*)xp;
        ax.q[1] = *(const uint4*)(xp + 8);
        v16bf bg = *(const v16bf*)wgp;
        v16bf bu = *(const v16bf*)wup;
        g = __builtin_amdgcn_wmma_f32_16x16x32_bf16(false, ax.v, false, bg,
                                                    (short)0, g, false, false);
        u = __builtin_amdgcn_wmma_f32_16x16x32_bf16(false, ax.v, false, bu,
                                                    (short)0, u, false, false);
        xp += 512; wgp += 512; wup += 512;
      }
      __syncthreads();                 // previous chunk's Hs fully consumed
      // silu(g)*u * comb -> bf16 H in LDS (row-major, padded stride)
#pragma unroll
      for (int i = 0; i < 8; i++) {
        float gv = g[i], uv = u[i];
        float h = (gv / (1.0f + __expf(-gv))) * uv * cw[i];
        int r = mhalf * 16 + i + 8 * rowSel;
        int c = wq * 16 + nIdx;
        Hs[r * HSTR + c] = f2bf(h);
      }
      __syncthreads();
      // ---- stage 2: down projection, accumulate into yacc ----
#pragma unroll
      for (int kt2 = 0; kt2 < 4; kt2++) {
        const int r  = mhalf * 16 + nIdx;          // A-frag: lane holds row lane&15
        const int cb = kt2 * 32 + (rowSel << 3);   // fragment K runs are contiguous
        union { uint4 q[2]; v16bf v; } af;
        af.q[0] = *(const uint4*)&Hs[r * HSTR + cb];
        af.q[1] = *(const uint4*)&Hs[r * HSTR + cb + 16];
        const int kIdx = ch * 4 + kt2;             // 0..47 k-tile along F
#pragma unroll
        for (int j = 0; j < 6; j++) {
          int nt = wq * 6 + j;                     // output n-tile 0..47
          v16bf bd = *(const v16bf*)(wd + ((size_t)nt * FK + kIdx) * 512 + lane * 16);
          yacc[j] = __builtin_amdgcn_wmma_f32_16x16x32_bf16(false, af.v, false, bd,
                                                            (short)0, yacc[j],
                                                            false, false);
        }
      }
    }
  }
  // ---- store fp32 partial output (each element written exactly once) ----
  float* po = part + (size_t)grp * T_TOK * DIM;
#pragma unroll
  for (int j = 0; j < 6; j++)
#pragma unroll
    for (int i = 0; i < 8; i++)
      po[(size_t)(tokBase + i + 8 * rowSel) * DIM + (wq * 6 + j) * 16 + nIdx] =
          yacc[j][i];
}

// =====================================================================
// Sum the 3 expert-group partials into the final output.
// =====================================================================
__global__ void moe_combine(const float* __restrict__ part,
                            float* __restrict__ out) {
  size_t i = (size_t)blockIdx.x * blockDim.x + threadIdx.x;
  const size_t n = (size_t)T_TOK * DIM;
  if (i < n)
    out[i] = part[i] + part[i + n] + part[i + 2 * n];
}

// =====================================================================
// launch
// =====================================================================
extern "C" void kernel_launch(void* const* d_in, const int* in_sizes, int n_in,
                              void* d_out, int out_size, void* d_ws, size_t ws_size,
                              hipStream_t stream) {
  const float* x    = (const float*)d_in[0];   // (T, D)
  const float* wr   = (const float*)d_in[1];   // (E, D)
  const float* bias = (const float*)d_in[2];   // (E,)
  const float* Wg   = (const float*)d_in[3];   // (E, F, D)
  const float* Wu   = (const float*)d_in[4];   // (E, F, D)
  const float* Wd   = (const float*)d_in[5];   // (E, D, F)
  const float* sg   = (const float*)d_in[6];   // (F, D)
  const float* su   = (const float*)d_in[7];   // (F, D)
  const float* sd   = (const float*)d_in[8];   // (D, F)
  float* out = (float*)d_out;

  // workspace layout (256B aligned)
  char* ws = (char*)d_ws;
  size_t off = 0;
  auto take = [&](size_t bytes) {
    char* p = ws + off;
    off = (off + bytes + 255) & ~(size_t)255;
    return p;
  };
  float* comb = (float*)take((size_t)T_TOK * 9 * sizeof(float));
  float* part = (float*)take((size_t)NGRP * T_TOK * DIM * sizeof(float));
  u16* xb  = (u16*)take((size_t)T_TOK * DIM * 2);
  u16* wgb = (u16*)take((size_t)NEXP * FF * DIM * 2);
  u16* wub = (u16*)take((size_t)NEXP * FF * DIM * 2);
  u16* wdb = (u16*)take((size_t)NEXP * DIM * FF * 2);
  u16* sgb = (u16*)take((size_t)FF * DIM * 2);
  u16* sub = (u16*)take((size_t)FF * DIM * 2);
  u16* sdb = (u16*)take((size_t)DIM * FF * 2);
  (void)ws_size; (void)n_in; (void)in_sizes; (void)out_size;

  // 1. router
  moe_router<<<T_TOK / 8, 256, 0, stream>>>(x, wr, bias, comb);

  // 2. pack activations & weights into bf16 WMMA fragment layout
  auto packGrid = [](int nTiles) { return (nTiles + 7) / 8; };
  moe_pack_frag<<<packGrid((T_TOK / 16) * DK), 256, 0, stream>>>(x, xb, T_TOK, DIM);
  moe_pack_frag<<<packGrid((NEXP * FF / 16) * DK), 256, 0, stream>>>(Wg, wgb, NEXP * FF, DIM);
  moe_pack_frag<<<packGrid((NEXP * FF / 16) * DK), 256, 0, stream>>>(Wu, wub, NEXP * FF, DIM);
  moe_pack_frag<<<packGrid((NEXP * DIM / 16) * FK), 256, 0, stream>>>(Wd, wdb, NEXP * DIM, FF);
  moe_pack_frag<<<packGrid((FF / 16) * DK), 256, 0, stream>>>(sg, sgb, FF, DIM);
  moe_pack_frag<<<packGrid((FF / 16) * DK), 256, 0, stream>>>(su, sub, FF, DIM);
  moe_pack_frag<<<packGrid((DIM / 16) * FK), 256, 0, stream>>>(sd, sdb, DIM, FF);

  // 3. fused MoE FFN: 3 expert groups in parallel -> partial buffers
  moe_main<<<dim3(T_TOK / 32, NGRP), 512, 0, stream>>>(
      xb, wgb, wub, wdb, sgb, sub, sdb, comb, part);

  // 4. combine partials
  const int n = T_TOK * DIM;
  moe_combine<<<(n + 255) / 256, 256, 0, stream>>>(part, out);
}
